// EaRL_51634096833093
// MI455X (gfx1250) — matrix operand order repaired and checked by reading, not compiled, not run on hardware
//
#include <hip/hip_runtime.h>
#include <math.h>

// ---------------------------------------------------------------------------
// Hetero GraphSAGE forward for gfx1250 (MI455X).
// Dense GEMMs use v_wmma_f32_16x16x32_bf16 (f32 accum); edge aggregation uses
// wave-per-edge global float atomics (memory-bound part of the workload).
// ---------------------------------------------------------------------------

typedef __attribute__((ext_vector_type(16))) __bf16 v16bf;
typedef __attribute__((ext_vector_type(8)))  float  v8f;

#define GF_ACCUM 1
#define GF_BIAS  2
#define GF_RELU  4

__device__ __forceinline__ unsigned short f2bf(float f) {
  unsigned u = __float_as_uint(f);
  u = u + 0x7FFFu + ((u >> 16) & 1u);   // round-to-nearest-even
  return (unsigned short)(u >> 16);
}

// ---------------- generic tiled WMMA GEMM:  C = [accum] + A(MxK,f32) * W(KxN,bf16) [+bias] [relu]
__global__ __launch_bounds__(128)
void k_gemm_bf16(const float* __restrict__ A, const unsigned short* __restrict__ W,
                 const float* __restrict__ bias, float* __restrict__ C,
                 int M, int K, int N, int flags)
{
  __shared__ __attribute__((aligned(16))) unsigned short sA [64 * 34];  // [row][k], pad 34
  __shared__ __attribute__((aligned(16))) unsigned short sBt[64 * 34];  // [col][k], pad 34

  const int tid   = threadIdx.x;
  const int lane  = tid & 31;
  const int wave  = tid >> 5;
  const int m     = lane & 15;
  const int hi    = lane >> 4;
  const int mBase = blockIdx.y * 64;
  const int nBase = blockIdx.x * 64;
  const int m0    = wave * 16;

  v8f acc[4] = {};

  for (int kb = 0; kb < K; kb += 32) {
    // A tile 64x32, f32 -> bf16
    for (int i = tid; i < 64 * 32; i += 128) {
      int r = i >> 5, k = i & 31;
      int gr_ = mBase + r, gk = kb + k;
      float v = (gr_ < M && gk < K) ? A[(long)gr_ * K + gk] : 0.0f;
      sA[r * 34 + k] = f2bf(v);
    }
    // W tile 32x64, stored transposed [n][k]
    for (int i = tid; i < 32 * 64; i += 128) {
      int k = i >> 6, n = i & 63;
      int gk = kb + k, gn = nBase + n;
      unsigned short v = (gk < K && gn < N) ? W[(long)gk * N + gn] : (unsigned short)0;
      sBt[n * 34 + k] = v;
    }
    __syncthreads();

    // A fragment: lane(m,hi); element pair j -> K = base(j) + hi*8, base = j<4 ? 2j : 16+2(j-4)
    union { v16bf v; unsigned int u[8]; } af;
#pragma unroll
    for (int j = 0; j < 8; j++) {
      int kk = ((j < 4) ? (2 * j) : (16 + 2 * (j - 4))) + hi * 8;
      af.u[j] = *(const unsigned int*)&sA[(m0 + m) * 34 + kk];
    }
#pragma unroll
    for (int t = 0; t < 4; t++) {
      // B fragment: lane holds column n0+m, element pair j -> K = base(j) + hi*8 (same as A)
      union { v16bf v; unsigned int u[8]; } bf_;
      int col = t * 16 + m;
#pragma unroll
      for (int j = 0; j < 8; j++) {
        int kk = ((j < 4) ? (2 * j) : (16 + 2 * (j - 4))) + hi * 8;
        bf_.u[j] = *(const unsigned int*)&sBt[col * 34 + kk];
      }
      acc[t] = __builtin_amdgcn_wmma_f32_16x16x32_bf16(
          false, af.v, false, bf_.v, (short)0, acc[t], false, false);
    }
    __syncthreads();
  }

#pragma unroll
  for (int t = 0; t < 4; t++) {
#pragma unroll
    for (int i = 0; i < 8; i++) {
      int row = mBase + m0 + i + 8 * hi;
      int col = nBase + t * 16 + m;
      if (row < M && col < N) {
        float v = acc[t][i];
        if (flags & GF_ACCUM) v += C[(long)row * N + col];
        if (flags & GF_BIAS)  v += bias[col];
        if (flags & GF_RELU)  v = fmaxf(v, 0.0f);
        C[(long)row * N + col] = v;
      }
    }
  }
}

// ---------------- small utility kernels ----------------
__global__ void k_zero(float* __restrict__ p, long n) {
  long i = (long)blockIdx.x * blockDim.x + threadIdx.x;
  long s = (long)gridDim.x * blockDim.x;
  for (; i < n; i += s) p[i] = 0.0f;
}

__global__ void k_copy(const float* __restrict__ a, float* __restrict__ b, long n) {
  long i = (long)blockIdx.x * blockDim.x + threadIdx.x;
  long s = (long)gridDim.x * blockDim.x;
  for (; i < n; i += s) b[i] = a[i];
}

__global__ void k_f32_to_bf16(const float* __restrict__ src, unsigned short* __restrict__ dst, long n) {
  long i = (long)blockIdx.x * blockDim.x + threadIdx.x;
  long s = (long)gridDim.x * blockDim.x;
  for (; i < n; i += s) dst[i] = f2bf(src[i]);
}

// wave-per-edge gather + atomic scatter-add (128 feature floats, 4 per lane)
__global__ __launch_bounds__(256)
void k_edge_agg(const float* __restrict__ xsrc, const int* __restrict__ esrc,
                const int* __restrict__ edst, float* __restrict__ agg,
                float* __restrict__ cnt, int E)
{
  int widx = (int)(((long)blockIdx.x * blockDim.x + threadIdx.x) >> 5);
  int lane = threadIdx.x & 31;
  if (widx >= E) return;
  int s = esrc[widx];
  int d = edst[widx];
  const float* xp = xsrc + (long)s * 128;
  float*       ap = agg  + (long)d * 128;
#pragma unroll
  for (int k = 0; k < 4; k++) atomicAdd(&ap[lane + 32 * k], xp[lane + 32 * k]);
  if (lane == 0) atomicAdd(&cnt[d], 1.0f);
}

__global__ void k_mean(float* __restrict__ agg, const float* __restrict__ cnt, int Nrow) {
  int row = blockIdx.x;
  if (row >= Nrow) return;
  float inv = 1.0f / fmaxf(cnt[row], 1.0f);
  agg[(long)row * 128 + threadIdx.x] *= inv;
}

// cross-layer scatter: xs1[j] = newx[pos] where map0[pos]==map1[j] (maps sorted unique)
__global__ __launch_bounds__(128)
void k_cross_scatter(const float* __restrict__ newx, const int* __restrict__ map0,
                     const int* __restrict__ map1, float* __restrict__ xs1, int n0, int n1)
{
  int j = blockIdx.x;
  if (j >= n1) return;
  __shared__ int spos;
  if (threadIdx.x == 0) {
    int key = map1[j];
    int lo = 0, hip_ = n0 - 1, pos = -1;
    while (lo <= hip_) {
      int mid = (lo + hip_) >> 1;
      int v = map0[mid];
      if (v == key) { pos = mid; break; }
      if (v < key) lo = mid + 1; else hip_ = mid - 1;
    }
    spos = pos;
  }
  __syncthreads();
  int p = spos;
  if (p >= 0) xs1[(long)j * 128 + threadIdx.x] = newx[(long)p * 128 + threadIdx.x];
}

// fused output heads: p_drop = sigmoid(relu(g@W1+b1)@W2+b2), values = relu(g@V1+c1)@V2+c2
__global__ __launch_bounds__(256)
void k_heads(const float* __restrict__ g,
             const float* __restrict__ dw1, const float* __restrict__ db1,
             const float* __restrict__ dw2, const float* __restrict__ db2,
             const float* __restrict__ vw1, const float* __restrict__ vb1,
             const float* __restrict__ vw2, const float* __restrict__ vb2,
             float* __restrict__ out, int Nrow)
{
  int row = blockIdx.x;
  if (row >= Nrow) return;
  __shared__ float gr[128];
  __shared__ float red[256];
  int tid = threadIdx.x;
  if (tid < 128) gr[tid] = g[(long)row * 128 + tid];
  __syncthreads();

  float hd = db1[tid];
#pragma unroll 4
  for (int k = 0; k < 128; k++) hd += gr[k] * dw1[k * 256 + tid];
  hd = fmaxf(hd, 0.0f);
  red[tid] = hd * dw2[tid];
  __syncthreads();
  for (int s = 128; s > 0; s >>= 1) {
    if (tid < s) red[tid] += red[tid + s];
    __syncthreads();
  }
  if (tid == 0) {
    float z = red[0] + db2[0];
    out[(long)row * 2 + 0] = 1.0f / (1.0f + expf(-z));
  }
  __syncthreads();

  float hv = vb1[tid];
#pragma unroll 4
  for (int k = 0; k < 128; k++) hv += gr[k] * vw1[k * 256 + tid];
  hv = fmaxf(hv, 0.0f);
  red[tid] = hv * vw2[tid];
  __syncthreads();
  for (int s = 128; s > 0; s >>= 1) {
    if (tid < s) red[tid] += red[tid + s];
    __syncthreads();
  }
  if (tid == 0) out[(long)row * 2 + 1] = red[0] + vb2[0];
}

// ---------------------------------------------------------------------------
extern "C" void kernel_launch(void* const* d_in, const int* in_sizes, int n_in,
                              void* d_out, int out_size, void* d_ws, size_t ws_size,
                              hipStream_t stream)
{
  (void)in_sizes; (void)n_in; (void)out_size; (void)ws_size;
  static const int NT[6]    = {25000, 15000, 4000, 12000, 12000, 15000};
  static const int FEAT6[6] = {257, 129, 128, 128, 129, 128};
  static const int REL_S[19] = {2,0,2,1,0,1,3,3,1,3,1,3,4,5,0,5,1,0,1};
  static const int REL_D[19] = {0,2,1,2,1,0,3,1,3,1,3,4,3,0,5,1,5,1,0};
  static const int GENE_REL[6] = {2,4,7,9,15,17};  // relations with dst == gene
  const int E = 100000, H = 128, HID = 256;
  (void)FEAT6;

  // ---- inputs (setup_inputs dict order: x/map interleaved, then edges, per layer)
  const float* x0[6];  const int* map0[6];
  const float* x1[6];  const int* map1[6];
  for (int t = 0; t < 6; t++) {
    x0[t]   = (const float*)d_in[2 * t];
    map0[t] = (const int*)  d_in[2 * t + 1];
    x1[t]   = (const float*)d_in[13 + 2 * t];
    map1[t] = (const int*)  d_in[13 + 2 * t + 1];
  }
  const int* edges0 = (const int*)d_in[12];
  const int* edges1 = (const int*)d_in[25];
  const float* Wia = (const float*)d_in[26]; const float* bia = (const float*)d_in[27];
  const float* Wig = (const float*)d_in[28]; const float* big = (const float*)d_in[29];
  const float* Wip = (const float*)d_in[30]; const float* bip = (const float*)d_in[31];
  const float* cwl = (const float*)d_in[32]; const float* cbl = (const float*)d_in[33];
  const float* cwr = (const float*)d_in[34];
  const float* mw1 = (const float*)d_in[35]; const float* mb1 = (const float*)d_in[36];
  const float* mw2 = (const float*)d_in[37]; const float* mb2 = (const float*)d_in[38];
  const float* gdw1 = (const float*)d_in[39]; const float* gdb1 = (const float*)d_in[40];
  const float* gdw2 = (const float*)d_in[41]; const float* gdb2 = (const float*)d_in[42];
  const float* gvw1 = (const float*)d_in[43]; const float* gvb1 = (const float*)d_in[44];
  const float* gvw2 = (const float*)d_in[45]; const float* gvb2 = (const float*)d_in[46];

  // ---- workspace carve-out
  size_t off = 0;
  float* base = (float*)d_ws;
  auto fall = [&](size_t n) { float* p = base + off; off += n; return p; };
  float* xs0[6];  for (int t = 0; t < 6; t++) xs0[t]  = fall((size_t)NT[t] * H);
  float* xs1[6];  for (int t = 0; t < 6; t++) xs1[t]  = fall((size_t)NT[t] * H);
  float* out0[6]; for (int t = 0; t < 6; t++) out0[t] = fall((size_t)NT[t] * H);
  size_t out0_total = (size_t)(25000 + 15000 + 4000 + 12000 + 12000 + 15000) * H;
  float* out1g  = fall((size_t)NT[1] * H);
  float* hidden = fall((size_t)25000 * HID);
  float* agg    = fall((size_t)25000 * H);
  float* cnt    = fall((size_t)25000);
  unsigned short* bf = (unsigned short*)(base + off);
  size_t boff = 0;
  auto ball = [&](size_t n) { unsigned short* p = bf + boff; boff += n; return p; };
  unsigned short* wl_bf = ball((size_t)19 * H * H);
  unsigned short* wr_bf = ball((size_t)19 * H * H);
  unsigned short* w1_bf = ball((size_t)6 * H * HID);
  unsigned short* w2_bf = ball((size_t)6 * HID * H);
  unsigned short* wa_bf = ball((size_t)257 * H);
  unsigned short* wg_bf = ball((size_t)129 * H);
  unsigned short* wp_bf = ball((size_t)129 * H);

  auto gblk = [](long n) { long b = (n + 255) / 256; return (int)(b > 8192 ? 8192 : b); };
  auto conv = [&](const float* s, unsigned short* d, long n) {
    k_f32_to_bf16<<<gblk(n), 256, 0, stream>>>(s, d, n);
  };
  auto gemm = [&](const float* A, const unsigned short* W, const float* bias, float* C,
                  int M, int K, int N, int flags) {
    dim3 g((N + 63) / 64, (M + 63) / 64);
    k_gemm_bf16<<<g, 128, 0, stream>>>(A, W, bias, C, M, K, N, flags);
  };

  // ---- 1) weight conversion (cheap; recomputed every call for determinism)
  conv(cwl, wl_bf, (long)19 * H * H);
  conv(cwr, wr_bf, (long)19 * H * H);
  conv(mw1, w1_bf, (long)6 * H * HID);
  conv(mw2, w2_bf, (long)6 * HID * H);
  conv(Wia, wa_bf, (long)257 * H);
  conv(Wig, wg_bf, (long)129 * H);
  conv(Wip, wp_bf, (long)129 * H);

  // ---- 2) input linears / pass-throughs for both layer snapshots
  const float* const* xin[2] = {x0, x1};
  float* const* xsl[2] = {xs0, xs1};
  for (int L = 0; L < 2; L++) {
    gemm(xin[L][0], wa_bf, bia, xsl[L][0], NT[0], 257, H, GF_BIAS);
    gemm(xin[L][1], wg_bf, big, xsl[L][1], NT[1], 129, H, GF_BIAS);
    gemm(xin[L][4], wp_bf, bip, xsl[L][4], NT[4], 129, H, GF_BIAS);
    for (int t : {2, 3, 5})
      k_copy<<<gblk((long)NT[t] * H), 256, 0, stream>>>(xin[L][t], xsl[L][t], (long)NT[t] * H);
  }

  // ---- 3) layer-0 hetero conv (all 19 relations, sum-aggregated per dst type)
  k_zero<<<gblk((long)out0_total), 256, 0, stream>>>(out0[0], (long)out0_total);
  auto relconv = [&](int r, const int* edges, float* const* xs, float* C) {
    int s = REL_S[r], d = REL_D[r];
    k_zero<<<gblk((long)NT[d] * H), 256, 0, stream>>>(agg, (long)NT[d] * H);
    k_zero<<<gblk((long)NT[d]), 256, 0, stream>>>(cnt, (long)NT[d]);
    const int* esrc = edges + (long)r * 2 * E;
    const int* edst = esrc + E;
    k_edge_agg<<<(E + 7) / 8, 256, 0, stream>>>(xs[s], esrc, edst, agg, cnt, E);
    k_mean<<<NT[d], 128, 0, stream>>>(agg, cnt, NT[d]);
    gemm(agg,   wl_bf + (long)r * H * H, cbl + (long)r * H, C, NT[d], H, H, GF_ACCUM | GF_BIAS);
    gemm(xs[d], wr_bf + (long)r * H * H, nullptr,           C, NT[d], H, H, GF_ACCUM);
  };
  for (int r = 0; r < 19; r++) relconv(r, edges0, xs0, out0[REL_D[r]]);

  // ---- 4) per-type layer MLP, then cross-layer map scatter into xs1
  for (int t = 0; t < 6; t++) {
    gemm(out0[t], w1_bf + (long)t * H * HID, mb1 + (long)t * HID, hidden, NT[t], H, HID,
         GF_BIAS | GF_RELU);
    gemm(hidden, w2_bf + (long)t * HID * H, mb2 + (long)t * H, out0[t], NT[t], HID, H, GF_BIAS);
    k_cross_scatter<<<NT[t], 128, 0, stream>>>(out0[t], map0[t], map1[t], xs1[t], NT[t], NT[t]);
  }

  // ---- 5) layer-1 hetero conv: only gene-dst relations are consumed
  k_zero<<<gblk((long)NT[1] * H), 256, 0, stream>>>(out1g, (long)NT[1] * H);
  for (int i = 0; i < 6; i++) relconv(GENE_REL[i], edges1, xs1, out1g);

  // ---- 6) fused heads -> [N_gene, 2]
  k_heads<<<NT[1], 256, 0, stream>>>(out1g, gdw1, gdb1, gdw2, gdb2,
                                     gvw1, gvb1, gvw2, gvb2, (float*)d_out, NT[1]);
}
